// GraphAttentionNetwork_75582834475118
// MI455X (gfx1250) — compile-verified, matrix operand order checked by reading, not run on hardware
//
#include <hip/hip_runtime.h>
#include <hip/hip_bf16.h>

// ---- problem constants (match reference) ----
static constexpr int N_HEADS  = 8;
static constexpr int IN_DIM   = 256;
static constexpr int EMB_DIM  = 256;
static constexpr int KDIM     = 32;              // key/val dim per head
static constexpr int HK       = N_HEADS * KDIM;  // 256
static constexpr int Bc       = 16;
static constexpr int NQc      = 1024;
static constexpr int Gc       = 1024;

typedef __attribute__((ext_vector_type(16))) __bf16 v16bf;
typedef __attribute__((ext_vector_type(8)))  float  v8f;
typedef __attribute__((ext_vector_type(4)))  int    i4v;

__device__ __forceinline__ __bf16 f2bf(float x) { return (__bf16)x; }

#if defined(__gfx1250__) && __has_builtin(__builtin_amdgcn_global_load_async_to_lds_b128)
#define HAVE_ASYNC_LDS 1
#else
#define HAVE_ASYNC_LDS 0
#endif

#define AS_GLOBAL __attribute__((address_space(1)))
#define AS_LOCAL  __attribute__((address_space(3)))

__device__ __forceinline__ void wait_asynccnt0() {
#if defined(__gfx1250__)
#if __has_builtin(__builtin_amdgcn_s_wait_asynccnt)
    __builtin_amdgcn_s_wait_asynccnt(0);
#else
    asm volatile("s_wait_asynccnt 0" ::: "memory");
#endif
#endif
}

// WMMA 16x16x32 bf16 operand layout (ISA 7.12.2), lane = (hf, lm):
//   A: row = lm; element j -> K = hf*8+j (j<8), 16+hf*8+(j-8) (j>=8)  [contiguous 8-groups]
//   B: mirrors A with lm = N column
//   C/D: lane holds column n = lm; vgpr r holds row m = hf*8 + r

// ============================================================
// Kernel 0: transpose weights to bf16, B-operand friendly layout.
//   WqT/WkT/WvT: [col = h*32+k][i]   (256 x 256)
//   WoT:         [col = e][r = h*32+k] (256 x 256)
// ============================================================
__global__ __launch_bounds__(256) void gan_prep_w_kernel(
    const float* __restrict__ Wq, const float* __restrict__ Wk,
    const float* __restrict__ Wv, const float* __restrict__ Wo,
    __bf16* __restrict__ WqT, __bf16* __restrict__ WkT,
    __bf16* __restrict__ WvT, __bf16* __restrict__ WoT)
{
    const int idx = blockIdx.x * 256 + threadIdx.x;   // 0..65535
    const int which = blockIdx.y;                     // 0..3
    if (which < 3) {
        const float* W = (which == 0) ? Wq : (which == 1) ? Wk : Wv;
        __bf16* WT = (which == 0) ? WqT : (which == 1) ? WkT : WvT;
        const int c = idx >> 8, i = idx & 255;        // col, input row
        const int hd = c >> 5, kk = c & 31;
        WT[idx] = f2bf(W[(size_t)hd * IN_DIM * KDIM + (size_t)i * KDIM + kk]);
    } else {
        const int e = idx >> 8, r = idx & 255;
        WoT[idx] = f2bf(Wo[(size_t)r * EMB_DIM + e]);
    }
}

// ============================================================
// Kernel 1: fused QKV projections. One wave = one 16-row M tile x 8 N tiles.
// role 0: Qp = q@Wq (row-major [B*NQ,256])
// role 1: Kp = h@Wk (row-major)
// role 2: VpT = (h@Wv) transposed -> [b, h, vdim, g]
// ============================================================
__global__ __launch_bounds__(256) void gan_proj_kernel(
    const float* __restrict__ qin, const float* __restrict__ hin,
    const __bf16* __restrict__ WqT, const __bf16* __restrict__ WkT,
    const __bf16* __restrict__ WvT,
    __bf16* __restrict__ Qp, __bf16* __restrict__ Kp, __bf16* __restrict__ VpT)
{
    const int wv = threadIdx.x >> 5;
    const int lane = threadIdx.x & 31;
    const int hf = lane >> 4;
    const int lm = lane & 15;
    const int role = blockIdx.z;

    const float* X;
    const __bf16* WT;
    if (role == 0)      { X = qin; WT = WqT; }
    else if (role == 1) { X = hin; WT = WkT; }
    else                { X = hin; WT = WvT; }

    const int mTile   = (blockIdx.x * 8 + wv) * 16;  // 1024 M tiles
    const int colHalf = blockIdx.y * 128;            // two 128-col halves

    v8f acc[8];
    #pragma unroll
    for (int t = 0; t < 8; ++t) acc[t] = (v8f){0.f,0.f,0.f,0.f,0.f,0.f,0.f,0.f};

    for (int k0 = 0; k0 < IN_DIM; k0 += 32) {
        v16bf a;
        const float* arow = X + (size_t)(mTile + lm) * IN_DIM + k0;
        #pragma unroll
        for (int j = 0; j < 8; ++j) {
            a[j]     = f2bf(arow[hf*8 + j]);
            a[j + 8] = f2bf(arow[16 + hf*8 + j]);
        }
        #pragma unroll
        for (int t = 0; t < 8; ++t) {
            v16bf bop;
            const __bf16* wb = WT + (size_t)(colHalf + t*16 + lm) * IN_DIM + k0;
            #pragma unroll
            for (int j = 0; j < 8; ++j) { bop[j] = wb[hf*8 + j]; bop[j+8] = wb[16 + hf*8 + j]; }
            acc[t] = __builtin_amdgcn_wmma_f32_16x16x32_bf16(false, a, false, bop,
                                                             (short)0, acc[t], false, false);
        }
    }

    if (role < 2) {
        __bf16* O = (role == 0) ? Qp : Kp;
        const int orow = mTile + hf * 8;
        #pragma unroll
        for (int t = 0; t < 8; ++t) {
            const int col = colHalf + t*16 + lm;
            #pragma unroll
            for (int r = 0; r < 8; ++r)
                O[(size_t)(orow + r) * HK + col] = f2bf(acc[t][r]);
        }
    } else {
        // transposed V store: VpT[((b*8+hd)*32 + kk)*Gc + g], 8 consecutive g per lane
        const int grow = mTile + hf * 8;
        const int bb = grow >> 10, gg = grow & 1023;
        #pragma unroll
        for (int t = 0; t < 8; ++t) {
            const int col = colHalf + t*16 + lm;
            const int hd = col >> 5, kk = col & 31;
            __bf16* vt = VpT + ((size_t)((bb * N_HEADS + hd) * KDIM + kk)) * Gc + gg;
            #pragma unroll
            for (int r = 0; r < 8; ++r) vt[r] = f2bf(acc[t][r]);   // contiguous 16B
        }
    }
}

// ============================================================
// Kernel 2: flash attention. Block = 4 waves sharing one (b, head); each wave
// owns a 16-query tile. K/V chunks (32 keys) staged once per block into LDS
// via gfx1250 async load-to-LDS. mask (67 MB) is read once per head (8x in total).
// ============================================================
__global__ __launch_bounds__(128) void gan_attn_kernel(
    const float* __restrict__ mask,
    const __bf16* __restrict__ Qp, const __bf16* __restrict__ Kp,
    const __bf16* __restrict__ VpT, __bf16* __restrict__ Hd)
{
    __shared__ __align__(16) __bf16 Ks[32 * 32];        // [key][kdim]
    __shared__ __align__(16) __bf16 Vs[32 * 32];        // [vdim][key]
    __shared__ __align__(16) __bf16 Pbuf[4][16 * 32];   // per-wave P transpose

    const int wv   = threadIdx.x >> 5;
    const int lane = threadIdx.x & 31;
    const int hf   = lane >> 4;
    const int lm   = lane & 15;

    // block decode: 2048 blocks = b(16) * hd(8) * qgrp(16)
    const int qgrp = blockIdx.x & 15;
    const int hd   = (blockIdx.x >> 4) & 7;
    const int b    = blockIdx.x >> 7;
    const int qBase = qgrp * 64 + wv * 16;
    const float normf = 0.17677669529663687f;           // 1/sqrt(32)

    // cooperative-copy indices: 128 threads x 16B = 2KB per buffer
    const int crow = threadIdx.x >> 2;
    const int cseg = (threadIdx.x & 3) << 3;

    // Q tile as A operand (load once)
    v16bf A;
    {
        const __bf16* qrow = Qp + (size_t)(b * NQc + qBase + lm) * HK + hd * KDIM;
        #pragma unroll
        for (int j = 0; j < 8; ++j) { A[j] = qrow[hf*8 + j]; A[j+8] = qrow[16 + hf*8 + j]; }
    }

    v8f O0 = {0.f,0.f,0.f,0.f,0.f,0.f,0.f,0.f};
    v8f O1 = {0.f,0.f,0.f,0.f,0.f,0.f,0.f,0.f};
    float mrow[8], lrow[8];
    #pragma unroll
    for (int r = 0; r < 8; ++r) { mrow[r] = -1e30f; lrow[r] = 0.f; }

    const float* mbase = mask + (size_t)b * NQc * Gc + (size_t)(qBase + hf*8) * Gc + lm;
    const __bf16* kgbase = Kp + (size_t)(b * Gc + crow) * HK + hd * KDIM + cseg;
    const __bf16* vgbase = VpT + ((size_t)((b * N_HEADS + hd) * KDIM + crow)) * Gc + cseg;

    for (int g0 = 0; g0 < Gc; g0 += 32) {
        // ---- stage K (32x32 bf16) and V^T (32x32 bf16) chunks into LDS ----
        {
            const __bf16* kg = kgbase + (size_t)g0 * HK;
            const __bf16* vg = vgbase + g0;
            __bf16* kl = Ks + crow * 32 + cseg;
            __bf16* vl = Vs + crow * 32 + cseg;
#if HAVE_ASYNC_LDS
            __builtin_amdgcn_global_load_async_to_lds_b128(
                (AS_GLOBAL i4v*)kg, (AS_LOCAL i4v*)kl, 0, 0);
            __builtin_amdgcn_global_load_async_to_lds_b128(
                (AS_GLOBAL i4v*)vg, (AS_LOCAL i4v*)vl, 0, 0);
            wait_asynccnt0();
#else
            *(uint4*)kl = *(const uint4*)kg;
            *(uint4*)vl = *(const uint4*)vg;
#endif
        }
        __syncthreads();

        // ---- scores: two 16-key B operands from LDS ----
        v16bf K0, K1;
        {
            const __bf16* kr0 = Ks + lm * 32;
            const __bf16* kr1 = Ks + (16 + lm) * 32;
            #pragma unroll
            for (int j = 0; j < 8; ++j) {
                K0[j] = kr0[hf*8 + j]; K0[j+8] = kr0[16 + hf*8 + j];
                K1[j] = kr1[hf*8 + j]; K1[j+8] = kr1[16 + hf*8 + j];
            }
        }
        const v8f Z = {0.f,0.f,0.f,0.f,0.f,0.f,0.f,0.f};
        v8f S0 = __builtin_amdgcn_wmma_f32_16x16x32_bf16(false, A, false, K0, (short)0, Z, false, false);
        v8f S1 = __builtin_amdgcn_wmma_f32_16x16x32_bf16(false, A, false, K1, (short)0, Z, false, false);

        // ---- additive mask + online softmax ----
        const float* mp = mbase + g0;
        __builtin_prefetch((const void*)(mp + 32), 0, 0);   // next mask chunk
        float p0[8], p1[8];
        #pragma unroll
        for (int r = 0; r < 8; ++r) {
            float s0 = S0[r] * normf + mp[(size_t)r * Gc];
            float s1 = S1[r] * normf + mp[(size_t)r * Gc + 16];
            float mx = fmaxf(s0, s1);
            mx = fmaxf(mx, __shfl_xor(mx, 1, 16));
            mx = fmaxf(mx, __shfl_xor(mx, 2, 16));
            mx = fmaxf(mx, __shfl_xor(mx, 4, 16));
            mx = fmaxf(mx, __shfl_xor(mx, 8, 16));
            const float nm = fmaxf(mrow[r], mx);
            const float sc = __expf(mrow[r] - nm);
            p0[r] = __expf(s0 - nm);
            p1[r] = __expf(s1 - nm);
            float rs = p0[r] + p1[r];
            rs += __shfl_xor(rs, 1, 16);
            rs += __shfl_xor(rs, 2, 16);
            rs += __shfl_xor(rs, 4, 16);
            rs += __shfl_xor(rs, 8, 16);
            lrow[r] = lrow[r] * sc + rs;
            mrow[r] = nm;
            O0[r] *= sc; O1[r] *= sc;
        }

        // ---- transpose P (D layout -> A layout) through per-wave LDS ----
        {
            __bf16* myl = Pbuf[wv];
            #pragma unroll
            for (int r = 0; r < 8; ++r) {
                myl[(hf*8 + r) * 32 + lm]      = f2bf(p0[r]);
                myl[(hf*8 + r) * 32 + 16 + lm] = f2bf(p1[r]);
            }
            asm volatile("s_wait_dscnt 0" ::: "memory");    // wave-local RAW fence
        }
        v16bf P;
        {
            const __bf16* prow = Pbuf[wv] + lm * 32;
            #pragma unroll
            for (int j = 0; j < 8; ++j) { P[j] = prow[hf*8 + j]; P[j+8] = prow[16 + hf*8 + j]; }
        }

        // ---- O += P @ V : two vdim-tile B operands from LDS (contiguous) ----
        v16bf V0, V1;
        {
            const __bf16* vr0 = Vs + lm * 32;
            const __bf16* vr1 = Vs + (16 + lm) * 32;
            #pragma unroll
            for (int j = 0; j < 8; ++j) {
                V0[j] = vr0[hf*8 + j]; V0[j+8] = vr0[16 + hf*8 + j];
                V1[j] = vr1[hf*8 + j]; V1[j+8] = vr1[16 + hf*8 + j];
            }
        }
        O0 = __builtin_amdgcn_wmma_f32_16x16x32_bf16(false, P, false, V0, (short)0, O0, false, false);
        O1 = __builtin_amdgcn_wmma_f32_16x16x32_bf16(false, P, false, V1, (short)0, O1, false, false);

        __syncthreads();   // all waves done with Ks/Vs before next stage
    }

    // ---- normalize and store heads (bf16, row-major [B*NQ, H*K]) ----
    __bf16* orow = Hd + (size_t)(b * NQc + qBase + hf*8) * HK + hd * KDIM + lm;
    #pragma unroll
    for (int r = 0; r < 8; ++r) {
        const float inv = 1.0f / lrow[r];
        orow[(size_t)r * HK]      = f2bf(O0[r] * inv);
        orow[(size_t)r * HK + 16] = f2bf(O1[r] * inv);
    }
}

// ============================================================
// Kernel 3: output projection. heads(16384x256) bf16 @ W_out -> f32 out.
// One wave = one 16-row M tile x 8 N tiles; B operand from pre-transposed WoT.
// ============================================================
__global__ __launch_bounds__(256) void gan_outproj_kernel(
    const __bf16* __restrict__ Hd, const __bf16* __restrict__ WoT,
    float* __restrict__ out)
{
    const int wv = threadIdx.x >> 5;
    const int lane = threadIdx.x & 31;
    const int hf = lane >> 4;
    const int lm = lane & 15;

    const int mTile   = (blockIdx.x * 8 + wv) * 16;
    const int colHalf = blockIdx.y * 128;

    v8f acc[8];
    #pragma unroll
    for (int t = 0; t < 8; ++t) acc[t] = (v8f){0.f,0.f,0.f,0.f,0.f,0.f,0.f,0.f};

    for (int k0 = 0; k0 < HK; k0 += 32) {
        v16bf a;
        const __bf16* arow = Hd + (size_t)(mTile + lm) * HK + k0;
        #pragma unroll
        for (int j = 0; j < 8; ++j) { a[j] = arow[hf*8 + j]; a[j+8] = arow[16 + hf*8 + j]; }
        #pragma unroll
        for (int t = 0; t < 8; ++t) {
            v16bf bop;
            const __bf16* wb = WoT + (size_t)(colHalf + t*16 + lm) * HK + k0;
            #pragma unroll
            for (int j = 0; j < 8; ++j) { bop[j] = wb[hf*8 + j]; bop[j+8] = wb[16 + hf*8 + j]; }
            acc[t] = __builtin_amdgcn_wmma_f32_16x16x32_bf16(false, a, false, bop,
                                                             (short)0, acc[t], false, false);
        }
    }
    const int orow = mTile + hf * 8;
    #pragma unroll
    for (int t = 0; t < 8; ++t) {
        float* op = out + (size_t)orow * EMB_DIM + colHalf + t*16 + lm;
        #pragma unroll
        for (int r = 0; r < 8; ++r) op[(size_t)r * EMB_DIM] = acc[t][r];
    }
}

// ============================================================
extern "C" void kernel_launch(void* const* d_in, const int* in_sizes, int n_in,
                              void* d_out, int out_size, void* d_ws, size_t ws_size,
                              hipStream_t stream) {
    const float* q    = (const float*)d_in[0];
    const float* h    = (const float*)d_in[1];
    const float* mask = (const float*)d_in[2];
    const float* Wq   = (const float*)d_in[3];
    const float* Wk   = (const float*)d_in[4];
    const float* Wv   = (const float*)d_in[5];
    const float* Wo   = (const float*)d_in[6];
    float* out = (float*)d_out;

    const size_t MAT = (size_t)Bc * NQc * HK * sizeof(__bf16);   // 8 MiB
    const size_t WSZ = (size_t)256 * 256 * sizeof(__bf16);       // 128 KiB
    char* ws = (char*)d_ws;
    __bf16* Qp  = (__bf16*)(ws);
    __bf16* Kp  = (__bf16*)(ws + MAT);
    __bf16* VpT = (__bf16*)(ws + 2 * MAT);
    __bf16* Hd  = (__bf16*)(ws + 3 * MAT);
    __bf16* WqT = (__bf16*)(ws + 4 * MAT);
    __bf16* WkT = (__bf16*)(ws + 4 * MAT + WSZ);
    __bf16* WvT = (__bf16*)(ws + 4 * MAT + 2 * WSZ);
    __bf16* WoT = (__bf16*)(ws + 4 * MAT + 3 * WSZ);

    // 0) transpose/convert weights (tiny)
    gan_prep_w_kernel<<<dim3(256, 4, 1), 256, 0, stream>>>(
        Wq, Wk, Wv, Wo, WqT, WkT, WvT, WoT);

    // 1) QKV projections: 8 waves/block, each wave 16x128 of output
    gan_proj_kernel<<<dim3(128, 2, 3), 256, 0, stream>>>(
        q, h, WqT, WkT, WvT, Qp, Kp, VpT);

    // 2) flash attention: 2048 blocks x 4 waves (one (b,head,64q) per block)
    gan_attn_kernel<<<dim3(2048, 1, 1), 128, 0, stream>>>(
        mask, Qp, Kp, VpT, Hd);

    // 3) output projection
    gan_outproj_kernel<<<dim3(128, 2, 1), 256, 0, stream>>>(Hd, WoT, out);
}
